// HFullyConnectedTensorProduct_79224966742157
// MI455X (gfx1250) — matrix operand order repaired and checked by reading, not compiled, not run on hardware
//
#include <hip/hip_runtime.h>
#include <math.h>

// ---------------------------------------------------------------------------
// Fused e3nn fully-connected tensor product for MI455X (gfx1250, wave32).
// Every term is a fused "outer-product A (built in-register) x bf16 weight"
// GEMM on v_wmma_f32_16x16x32_bf16 with fp32 accumulation.
// N-blocking: each A fragment is reused by TWO WMMAs (two 16-col output
// tiles, two accumulators), halving VALU A-construction cost per WMMA so the
// matrix pipe is the bottleneck. Norm constants folded into bf16 weights.
// ---------------------------------------------------------------------------

typedef __attribute__((ext_vector_type(16))) __bf16 v16bf;
typedef __attribute__((ext_vector_type(8)))  float  v8f;

#define BATCH   8192
#define OUTW    512
#define S_STR   132   // 128 + 4 pad (floats) -> bank spread across rows
#define V_STR   68    // 64 + 4 pad

// --- B fragment: 32x16 bf16, lane = N column (lane%16); element e<8 holds
// K = 8*kh+e, element e>=8 holds K = 16+8*kh+(e-8) (kstart = 8*kh). ---
__device__ __forceinline__ v16bf load_bfrag(const __bf16* wrow, int kstart) {
  union { uint4 u4[2]; v16bf v; } f;
  f.u4[0] = *reinterpret_cast<const uint4*>(wrow + kstart);
  f.u4[1] = *reinterpret_cast<const uint4*>(wrow + kstart + 16);
  return f.v;
}

// --- A fragment: 16x32 bf16, lane = M row (lane%16); element e<8 holds
// K = 8*kh+e, element e>=8 holds K = 16+8*kh+(e-8). q points at the row
// already offset to the chunk base v0. ---
__device__ __forceinline__ v16bf fragA_mul(const float* __restrict__ q,
                                           int kh, float p) {
  const int o0 = 8 * kh, o1 = 16 + 8 * kh;
  v16bf r;
#pragma unroll
  for (int e = 0; e < 8; ++e) {
    r[e]     = (__bf16)(p * q[o0 + e]);
    r[e + 8] = (__bf16)(p * q[o1 + e]);
  }
  return r;
}

// A[m][(u,v)] = sum_i v1[u,i]*v2[v,i]  (dot channel, one WMMA pass)
__device__ __forceinline__ v16bf fragA_dot3(const float* __restrict__ q0,
                                            const float* __restrict__ q1,
                                            const float* __restrict__ q2,
                                            int kh, float p0, float p1, float p2) {
  const int o0 = 8 * kh, o1 = 16 + 8 * kh;
  v16bf r;
#pragma unroll
  for (int e = 0; e < 8; ++e) {
    r[e]     = (__bf16)(p0 * q0[o0 + e] + p1 * q1[o0 + e] + p2 * q2[o0 + e]);
    r[e + 8] = (__bf16)(p0 * q0[o1 + e] + p1 * q1[o1 + e] + p2 * q2[o1 + e]);
  }
  return r;
}

// A[m][(u,v)] = pi*qj[v] - pj*qi[v]  (cross channel, per output component)
__device__ __forceinline__ v16bf fragA_cross(const float* __restrict__ qj,
                                             const float* __restrict__ qi,
                                             int kh, float pi, float pj) {
  const int o0 = 8 * kh, o1 = 16 + 8 * kh;
  v16bf r;
#pragma unroll
  for (int e = 0; e < 8; ++e) {
    r[e]     = (__bf16)(pi * qj[o0 + e] - pj * qi[o0 + e]);
    r[e + 8] = (__bf16)(pi * qj[o1 + e] - pj * qi[o1 + e]);
  }
  return r;
}

#define WMMA_BF16(a, b, c) \
  __builtin_amdgcn_wmma_f32_16x16x32_bf16(false, (a), false, (b), (short)0, (c), false, false)

// ---------------------------------------------------------------------------
// Weight prep: W[u][v][w] (f32) -> Wt[w][u*V+v] (bf16) with norm folded in.
// Runs once per launch; weights are tiny (<8 MB bf16) and then L2-resident.
// ---------------------------------------------------------------------------
__global__ void prep_kernel(const float* __restrict__ src, __bf16* __restrict__ dst,
                            int Nw, float scale, int total) {
  int e = blockIdx.x * 256 + threadIdx.x;
  if (e >= total) return;
  int w  = e % Nw;
  int uv = e / Nw;           // = u*V + v  (src is (U,V,Nw) row-major)
  dst[(size_t)w * (size_t)(total / Nw) + uv] = (__bf16)(src[e] * scale);
}

// ---------------------------------------------------------------------------
// Main fused kernel: grid.x = batch tile (16 rows); blockIdx.y*8 + wave = one
// of 16 jobs, each owning TWO 16-col output tiles. 256 threads = 8 waves.
// ---------------------------------------------------------------------------
__global__ __launch_bounds__(256) void tp_kernel(
    const float* __restrict__ x1, const float* __restrict__ x2,
    const __bf16* __restrict__ w000t, const __bf16* __restrict__ w011t,
    const __bf16* __restrict__ w101t, const __bf16* __restrict__ w110t,
    const __bf16* __restrict__ w111t, float* __restrict__ out) {

  __shared__ float s1s[16 * S_STR];
  __shared__ float s2s[16 * S_STR];
  __shared__ float v1s[3][16 * V_STR];
  __shared__ float v2s[3][16 * V_STR];

  const int mbase = blockIdx.x * 16;
  const int tid   = threadIdx.x;

  // Stage 16 rows of x1/x2 into LDS, de-interleaving the l=1 components.
  for (int idx = tid; idx < 16 * 320; idx += 256) {
    int m = idx / 320, c = idx % 320;
    float a = x1[(size_t)(mbase + m) * 320 + c];
    float b = x2[(size_t)(mbase + m) * 320 + c];
    if (c < 128) {
      s1s[m * S_STR + c] = a;
      s2s[m * S_STR + c] = b;
    } else {
      int cc = c - 128, u = cc / 3, i = cc % 3;
      v1s[i][m * V_STR + u] = a;
      v2s[i][m * V_STR + u] = b;
    }
  }
  __syncthreads();

  const int wave = tid >> 5;
  const int lane = tid & 31;
  const int job  = blockIdx.y * 8 + wave;   // 0..15
  const int kh   = lane >> 4;               // K-half for A; K-group for B
  const int n    = lane & 15;               // A row m, B/C column n
  const int m    = n;
  const int bks  = 8 * kh;                  // B fragment K start inside chunk

  v8f acc0 = {};
  v8f acc1 = {};

  if (job < 4) {
    // ---- out0: w000 (s1 (x) s2) + w110 (v1 . v2), shared accumulators ----
    const int wb0 = job * 32, wb1 = wb0 + 16;
    const float* q = s2s + m * S_STR;
    const __bf16* wr0 = w000t + (size_t)(wb0 + n) * 16384;
    const __bf16* wr1 = w000t + (size_t)(wb1 + n) * 16384;
    for (int u = 0; u < 128; ++u) {
      float p = s1s[m * S_STR + u];
#pragma unroll
      for (int vc = 0; vc < 4; ++vc) {
        v16bf a  = fragA_mul(q + vc * 32, kh, p);
        v16bf b0 = load_bfrag(wr0 + u * 128 + vc * 32, bks);
        v16bf b1 = load_bfrag(wr1 + u * 128 + vc * 32, bks);
        acc0 = WMMA_BF16(a, b0, acc0);
        acc1 = WMMA_BF16(a, b1, acc1);
      }
    }
    const float* q0 = v2s[0] + m * V_STR;
    const float* q1 = v2s[1] + m * V_STR;
    const float* q2 = v2s[2] + m * V_STR;
    const __bf16* wd0 = w110t + (size_t)(wb0 + n) * 4096;
    const __bf16* wd1 = w110t + (size_t)(wb1 + n) * 4096;
    for (int u = 0; u < 64; ++u) {
      float p0 = v1s[0][m * V_STR + u];
      float p1 = v1s[1][m * V_STR + u];
      float p2 = v1s[2][m * V_STR + u];
#pragma unroll
      for (int vc = 0; vc < 2; ++vc) {
        v16bf a  = fragA_dot3(q0 + vc * 32, q1 + vc * 32, q2 + vc * 32, kh, p0, p1, p2);
        v16bf b0 = load_bfrag(wd0 + u * 64 + vc * 32, bks);
        v16bf b1 = load_bfrag(wd1 + u * 64 + vc * 32, bks);
        acc0 = WMMA_BF16(a, b0, acc0);
        acc1 = WMMA_BF16(a, b1, acc1);
      }
    }
    const int row0 = mbase + kh * 8;
#pragma unroll
    for (int r = 0; r < 8; ++r) {
      out[(size_t)(row0 + r) * OUTW + wb0 + n] = acc0[r];
      out[(size_t)(row0 + r) * OUTW + wb1 + n] = acc1[r];
    }

  } else if (job < 10) {
    // ---- out1o component i: w011 (s1 (x) v2_i) + w101 (v1_i (x) s2) ----
    const int idx = job - 4, i = idx >> 1, wb0 = (idx & 1) * 32, wb1 = wb0 + 16;
    const float* qv = v2s[i] + m * V_STR;
    const __bf16* wa0 = w011t + (size_t)(wb0 + n) * 8192;
    const __bf16* wa1 = w011t + (size_t)(wb1 + n) * 8192;
    for (int u = 0; u < 128; ++u) {
      float p = s1s[m * S_STR + u];
#pragma unroll
      for (int vc = 0; vc < 2; ++vc) {
        v16bf a  = fragA_mul(qv + vc * 32, kh, p);
        v16bf b0 = load_bfrag(wa0 + u * 64 + vc * 32, bks);
        v16bf b1 = load_bfrag(wa1 + u * 64 + vc * 32, bks);
        acc0 = WMMA_BF16(a, b0, acc0);
        acc1 = WMMA_BF16(a, b1, acc1);
      }
    }
    const float* qs = s2s + m * S_STR;
    const __bf16* wb0p = w101t + (size_t)(wb0 + n) * 8192;
    const __bf16* wb1p = w101t + (size_t)(wb1 + n) * 8192;
    for (int u = 0; u < 64; ++u) {
      float p = v1s[i][m * V_STR + u];
#pragma unroll
      for (int vc = 0; vc < 4; ++vc) {
        v16bf a  = fragA_mul(qs + vc * 32, kh, p);
        v16bf b0 = load_bfrag(wb0p + u * 128 + vc * 32, bks);
        v16bf b1 = load_bfrag(wb1p + u * 128 + vc * 32, bks);
        acc0 = WMMA_BF16(a, b0, acc0);
        acc1 = WMMA_BF16(a, b1, acc1);
      }
    }
    const int row0 = mbase + kh * 8;
#pragma unroll
    for (int r = 0; r < 8; ++r) {
      out[(size_t)(row0 + r) * OUTW + 128 + 3 * (wb0 + n) + i] = acc0[r];
      out[(size_t)(row0 + r) * OUTW + 128 + 3 * (wb1 + n) + i] = acc1[r];
    }

  } else {
    // ---- out1e component k: w111 with cross product (v1 x v2)_k ----
    const int idx = job - 10, kc = idx >> 1, wb0 = (idx & 1) * 32, wb1 = wb0 + 16;
    const int i = (kc + 1) % 3, j = (kc + 2) % 3;   // eps[i][j][kc] = +1
    const float* qj = v2s[j] + m * V_STR;
    const float* qi = v2s[i] + m * V_STR;
    const __bf16* wc0 = w111t + (size_t)(wb0 + n) * 4096;
    const __bf16* wc1 = w111t + (size_t)(wb1 + n) * 4096;
    for (int u = 0; u < 64; ++u) {
      float pi = v1s[i][m * V_STR + u];
      float pj = v1s[j][m * V_STR + u];
#pragma unroll
      for (int vc = 0; vc < 2; ++vc) {
        v16bf a  = fragA_cross(qj + vc * 32, qi + vc * 32, kh, pi, pj);
        v16bf b0 = load_bfrag(wc0 + u * 64 + vc * 32, bks);
        v16bf b1 = load_bfrag(wc1 + u * 64 + vc * 32, bks);
        acc0 = WMMA_BF16(a, b0, acc0);
        acc1 = WMMA_BF16(a, b1, acc1);
      }
    }
    const int row0 = mbase + kh * 8;
#pragma unroll
    for (int r = 0; r < 8; ++r) {
      out[(size_t)(row0 + r) * OUTW + 320 + 3 * (wb0 + n) + kc] = acc0[r];
      out[(size_t)(row0 + r) * OUTW + 320 + 3 * (wb1 + n) + kc] = acc1[r];
    }
  }
}

// ---------------------------------------------------------------------------
extern "C" void kernel_launch(void* const* d_in, const int* in_sizes, int n_in,
                              void* d_out, int out_size, void* d_ws, size_t ws_size,
                              hipStream_t stream) {
  (void)in_sizes; (void)n_in; (void)out_size; (void)ws_size;
  const float* x1   = (const float*)d_in[0];
  const float* x2   = (const float*)d_in[1];
  const float* w000 = (const float*)d_in[2];
  const float* w011 = (const float*)d_in[3];
  const float* w101 = (const float*)d_in[4];
  const float* w110 = (const float*)d_in[5];
  const float* w111 = (const float*)d_in[6];
  float* out = (float*)d_out;

  __bf16* ws    = (__bf16*)d_ws;
  __bf16* w000t = ws;                               // 128 x 16384
  __bf16* w011t = w000t + (size_t)128 * 16384;      //  64 x  8192
  __bf16* w101t = w011t + (size_t)64 * 8192;        //  64 x  8192
  __bf16* w110t = w101t + (size_t)64 * 8192;        // 128 x  4096
  __bf16* w111t = w110t + (size_t)128 * 4096;       //  64 x  4096

  // Fold normalization constants into the weights.
  const float c000 = 1.0f / sqrtf(128.f * 128.f + 64.f * 64.f); // A_0E
  const float c110 = c000 / sqrtf(3.0f);                        // A_0E/sqrt3
  const float c011 = 1.0f / 128.0f;                             // A_1O/sqrt3
  const float c101 = 1.0f / 128.0f;
  const float c111 = 1.0f / (64.0f * sqrtf(2.0f));              // A_1E/sqrt6

  auto blocks = [](int total) { return (total + 255) / 256; };
  prep_kernel<<<blocks(128 * 128 * 128), 256, 0, stream>>>(w000, w000t, 128, c000, 128 * 128 * 128);
  prep_kernel<<<blocks(128 * 64 * 64),  256, 0, stream>>>(w011, w011t, 64,  c011, 128 * 64 * 64);
  prep_kernel<<<blocks(64 * 128 * 64),  256, 0, stream>>>(w101, w101t, 64,  c101, 64 * 128 * 64);
  prep_kernel<<<blocks(64 * 64 * 128),  256, 0, stream>>>(w110, w110t, 128, c110, 64 * 64 * 128);
  prep_kernel<<<blocks(64 * 64 * 64),   256, 0, stream>>>(w111, w111t, 64,  c111, 64 * 64 * 64);

  dim3 grid(BATCH / 16, 2);
  tp_kernel<<<grid, 256, 0, stream>>>(x1, x2, w000t, w011t, w101t, w110t, w111t, out);
}
